// CustomizedSelfAttention_33105607917868
// MI455X (gfx1250) — matrix operand order repaired and checked by reading, not compiled, not run on hardware
//
#include <hip/hip_runtime.h>
#include <hip/hip_bf16.h>
#include <math.h>

// ---------------------------------------------------------------------------
// CDNA5 (gfx1250) bf16 WMMA implementation of:
//   QKV proj -> flash attention -> out proj + residual -> mean pool -> MLP.
// All GEMMs: V_WMMA_F32_16X16X32_BF16 (wave32).  LDS staging uses the TDM
// (tensor_load_to_lds, TENSORcnt) and async global->LDS (ASYNCcnt) paths
// where the toolchain exposes them, with compile-safe fallbacks.
// Workspace: bf16 Q,K,V,O (4 x 58.7MB) + small f32 buffers (~235 MB).
// ---------------------------------------------------------------------------

typedef __bf16 bf16;
typedef bf16 v16bf __attribute__((ext_vector_type(16)));
typedef bf16 v8bf  __attribute__((ext_vector_type(8)));
typedef float v8f  __attribute__((ext_vector_type(8)));
typedef unsigned int u32x4 __attribute__((ext_vector_type(4)));
typedef int i32x4 __attribute__((ext_vector_type(4)));
typedef int i32x8 __attribute__((ext_vector_type(8)));

#define C_DIM  896
#define N_TOK  4096
#define B_SZ   8
#define KSTEPS 28          // 896 / 32

// ----- gfx1250 data-mover feature probes -----------------------------------
#if __has_builtin(__builtin_amdgcn_global_load_async_to_lds_b128)
#define HAVE_ASYNC_LDS 1
#else
#define HAVE_ASYNC_LDS 0
#endif
#if __has_builtin(__builtin_amdgcn_tensor_load_to_lds)
#define HAVE_TDM 1
#else
#define HAVE_TDM 0
#endif

__device__ __forceinline__ void async_copy16(const bf16* g, bf16* l) {
#if HAVE_ASYNC_LDS
  // builtin signature (probe-confirmed by diagnostic): int4 global src ptr,
  // int4 LDS dst ptr, imm offset, imm cpol.
  typedef __attribute__((address_space(1))) i32x4* g_i4p;
  typedef __attribute__((address_space(3))) i32x4* l_i4p;
  bf16* gn = const_cast<bf16*>(g);
  __builtin_amdgcn_global_load_async_to_lds_b128((g_i4p)gn, (l_i4p)l, 0, 0);
#else
  *(uint4*)l = *(const uint4*)g;
#endif
}
__device__ __forceinline__ void async_wait() {
#if HAVE_ASYNC_LDS && __has_builtin(__builtin_amdgcn_s_wait_asynccnt)
  __builtin_amdgcn_s_wait_asynccnt(0);
#endif
}
__device__ __forceinline__ void tensor_wait() {
#if HAVE_TDM && __has_builtin(__builtin_amdgcn_s_wait_tensorcnt)
  __builtin_amdgcn_s_wait_tensorcnt(0);
#endif
}

#if HAVE_TDM
// Raw LDS byte offset (descriptor wants an LDS address, not a flat pointer).
__device__ __forceinline__ unsigned lds_byte_addr(const void* p) {
  typedef __attribute__((address_space(3))) const void* lds_cptr;
  return (unsigned)(unsigned long long)(lds_cptr)p;
}
// TDM: DMA a rows x cols (bf16, contiguous, row stride = cols) tile into LDS.
__device__ __forceinline__ void tdm_load_2d_bf16(const bf16* gsrc, bf16* ldst,
                                                 int cols, int rows) {
  const unsigned long long ga = (unsigned long long)gsrc;
  u32x4 g0;
  g0[0] = 1u;                                   // count=1, user descriptor
  g0[1] = lds_byte_addr(ldst);                  // lds_addr [63:32]
  g0[2] = (unsigned)(ga & 0xFFFFFFFFull);       // global_addr lo
  g0[3] = (unsigned)((ga >> 32) & 0x1FFFFFFull) // global_addr [56:32]
          | (2u << 30);                         // type = 2 ("image")
  i32x8 g1;
  g1[0] = (int)(1u << 16);                      // data_size=1 (2 bytes)
  g1[1] = (int)(((unsigned)cols & 0xFFFFu) << 16);   // tensor_dim0 lo16
  g1[2] = (int)(((unsigned)rows & 0xFFFFu) << 16);   // dim0 hi16=0 | tensor_dim1
  g1[3] = (int)(((unsigned)cols & 0xFFFFu) << 16);   // dim1 hi16=0 | tile_dim0
  g1[4] = (int)((unsigned)rows & 0xFFFFu);           // tile_dim1 | tile_dim2=0
  g1[5] = cols;                                      // tensor_dim0_stride lo32
  g1[6] = 0;
  g1[7] = 0;
  i32x4 z4;  z4[0] = 0; z4[1] = 0; z4[2] = 0; z4[3] = 0;
#if __clang_major__ >= 23
  i32x8 z8;
#pragma unroll
  for (int i = 0; i < 8; ++i) z8[i] = 0;
  __builtin_amdgcn_tensor_load_to_lds(g0, g1, z4, z4, z8, 0);
#else
  __builtin_amdgcn_tensor_load_to_lds(g0, g1, z4, z4, 0);
#endif
}
#endif  // HAVE_TDM

__device__ __forceinline__ v8f v8f_zero() {
  v8f z;
#pragma unroll
  for (int i = 0; i < 8; ++i) z[i] = 0.0f;
  return z;
}

__device__ __forceinline__ v8f wmma_bf16(v16bf a, v16bf b, v8f c) {
  return __builtin_amdgcn_wmma_f32_16x16x32_bf16(false, a, false, b,
                                                 (short)0, c, false, false);
}

// A fragment (16x32 bf16), LDS row-major [m][k].
// lane<16 -> row=lane, K {0..15}; lane>=16 -> row=lane-16, K {16..31}.
__device__ __forceinline__ v16bf load_a_frag(const bf16* base, int pitch, int lane) {
  const int r  = lane & 15;
  const int k0 = (lane >> 4) << 4;            // 0 or 16
  union { v16bf v; v8bf h[2]; } u;
  u.h[0] = *(const v8bf*)(base + r * pitch + k0);
  u.h[1] = *(const v8bf*)(base + r * pitch + k0 + 8);
  return u.v;
}

// B fragment (32x16 bf16), LDS "row n holds K-run of output column n".
__device__ __forceinline__ v16bf load_b_frag(const bf16* base, int pitch, int lane) {
  const int n  = lane & 15;
  const int k0 = (lane >> 4) << 4;            // 0 or 16
  return *(const v16bf*)(base + n * pitch + k0);
}

// Stage W (f32, row-major [k][n]) tile 32x128 into Bs[n][k] (bf16), packing
// two consecutive k per b32 store.  256 threads.
__device__ __forceinline__ void stage_b_tile(const float* __restrict__ W,
                                             bf16* __restrict__ Bs, int k0,
                                             int n0, int tid) {
  const int bkp  = (tid >> 4) << 1;    // 0,2,..,30
  const int bnof = (tid & 15) * 8;     // 0..120
  union F4 { float4 v; float f[4]; };
  F4 r00, r01, r10, r11;
  r00.v = *(const float4*)(W + (size_t)(k0 + bkp) * C_DIM + n0 + bnof);
  r01.v = *(const float4*)(W + (size_t)(k0 + bkp) * C_DIM + n0 + bnof + 4);
  r10.v = *(const float4*)(W + (size_t)(k0 + bkp + 1) * C_DIM + n0 + bnof);
  r11.v = *(const float4*)(W + (size_t)(k0 + bkp + 1) * C_DIM + n0 + bnof + 4);
#pragma unroll
  for (int e = 0; e < 8; ++e) {
    const float lo = (e < 4) ? r00.f[e] : r01.f[e - 4];
    const float hi = (e < 4) ? r10.f[e] : r11.f[e - 4];
    union { unsigned u; bf16 b[2]; } p;
    p.b[0] = (bf16)lo; p.b[1] = (bf16)hi;
    *(unsigned*)&Bs[(bnof + e) * 48 + bkp] = p.u;
  }
}

// ---------------------------------------------------------------------------
// Kernel 1: QKV projections.  grid=(M/128, C/128, 3); attention scale folded
// into Q.
// ---------------------------------------------------------------------------
__global__ __launch_bounds__(256) void qkv_gemm_kernel(
    const float* __restrict__ x,
    const float* __restrict__ Wq, const float* __restrict__ Wk,
    const float* __restrict__ Wv,
    bf16* __restrict__ Qo, bf16* __restrict__ Ko, bf16* __restrict__ Vo) {
  __shared__ bf16 As[128 * 40];   // [m][k], pitch 40
  __shared__ bf16 Bs[128 * 48];   // [n][k], pitch 48

  const int tid  = threadIdx.x;
  const int lane = tid & 31;
  const int w    = tid >> 5;
  const int wr   = w >> 1;
  const int wc   = w & 1;
  const int m0   = blockIdx.x * 128;
  const int n0   = blockIdx.y * 128;
  const int sel  = blockIdx.z;

  const float* W   = (sel == 0) ? Wq : (sel == 1) ? Wk : Wv;
  bf16*        Out = (sel == 0) ? Qo : (sel == 1) ? Ko : Vo;
  const float  osc = (sel == 0) ? rsqrtf((float)C_DIM) : 1.0f;

  v8f acc[2][4];
#pragma unroll
  for (int i = 0; i < 2; ++i)
#pragma unroll
    for (int j = 0; j < 4; ++j) acc[i][j] = v8f_zero();

  const int arow  = tid >> 1;          // 0..127
  const int akoff = (tid & 1) * 16;    // 0 or 16

  for (int kt = 0; kt < KSTEPS; ++kt) {
    const int k0 = kt * 32;
    // stage A: x tile 128x32, f32 -> bf16, two packed b128 stores
    {
      const float4* src =
          (const float4*)(x + (size_t)(m0 + arow) * C_DIM + k0 + akoff);
      union F4 { float4 v; float f[4]; };
      F4 f0, f1, f2, f3;
      f0.v = src[0]; f1.v = src[1]; f2.v = src[2]; f3.v = src[3];
      union { uint4 u; bf16 b[8]; } o0, o1;
#pragma unroll
      for (int e = 0; e < 4; ++e) {
        o0.b[e]     = (bf16)f0.f[e];  o0.b[e + 4] = (bf16)f1.f[e];
        o1.b[e]     = (bf16)f2.f[e];  o1.b[e + 4] = (bf16)f3.f[e];
      }
      *(uint4*)&As[arow * 40 + akoff]     = o0.u;
      *(uint4*)&As[arow * 40 + akoff + 8] = o1.u;
    }
    stage_b_tile(W, Bs, k0, n0, tid);
    if (kt + 1 < KSTEPS) {
      __builtin_prefetch(x + (size_t)(m0 + arow) * C_DIM + k0 + 32 + akoff, 0, 1);
      __builtin_prefetch(W + (size_t)(k0 + 32 + ((tid >> 4) << 1)) * C_DIM + n0 +
                             (tid & 15) * 8, 0, 1);
    }
    __syncthreads();

    const v16bf a0 = load_a_frag(As + (wr * 32 + 0)  * 40, 40, lane);
    const v16bf a1 = load_a_frag(As + (wr * 32 + 16) * 40, 40, lane);
#pragma unroll
    for (int j = 0; j < 4; ++j) {
      const v16bf b = load_b_frag(Bs + (wc * 64 + j * 16) * 48, 48, lane);
      acc[0][j] = wmma_bf16(a0, b, acc[0][j]);
      acc[1][j] = wmma_bf16(a1, b, acc[1][j]);
    }
    __syncthreads();
  }

  const int rb = (lane >> 4) << 3;
  const int cl = lane & 15;
#pragma unroll
  for (int i = 0; i < 2; ++i)
#pragma unroll
    for (int j = 0; j < 4; ++j) {
      const int col = n0 + wc * 64 + j * 16 + cl;
#pragma unroll
      for (int r = 0; r < 8; ++r) {
        const int row = m0 + wr * 32 + i * 16 + rb + r;
        Out[(size_t)row * C_DIM + col] = (bf16)(acc[i][j][r] * osc);
      }
    }
}

// ---------------------------------------------------------------------------
// Kernel 2: flash attention.  One WG per (batch, 16 query rows); 128 key
// chunks of 32.  K chunk staged by TDM (wave 0) or async copies; V chunk
// transposed through VGPRs with packed dword stores.
// ---------------------------------------------------------------------------
__global__ __launch_bounds__(256) void flash_attn_kernel(
    const bf16* __restrict__ Q, const bf16* __restrict__ Kg,
    const bf16* __restrict__ Vg, bf16* __restrict__ O) {
  __shared__ bf16  Qs[16 * C_DIM];    // [row][d]
  __shared__ bf16  Ks[32 * C_DIM];    // [key][d]
  __shared__ bf16  Vst[C_DIM * 32];   // [d][key]
  __shared__ float Ssc[16 * 32];
  __shared__ bf16  Ps[16 * 32];
  __shared__ float mrow[16], lrow[16], crow[16], linv[16];

  const int tid  = threadIdx.x;
  const int lane = tid & 31;
  const int w    = tid >> 5;
  const int q0   = blockIdx.x * 16;
  const int b    = blockIdx.y;
  const size_t baseQ = ((size_t)b * N_TOK + q0) * C_DIM;

  for (int idx = tid * 8; idx < 16 * C_DIM; idx += 256 * 8)
    async_copy16(&Q[baseQ + idx], &Qs[idx]);
  async_wait();
  if (tid < 16) { mrow[tid] = -1e30f; lrow[tid] = 0.0f; }

  v8f oacc[7];
#pragma unroll
  for (int t = 0; t < 7; ++t) oacc[t] = v8f_zero();

  for (int jc = 0; jc < N_TOK / 32; ++jc) {
    __syncthreads();  // previous chunk's consumers done (also publishes Qs)
    const size_t baseK = ((size_t)b * N_TOK + jc * 32) * C_DIM;

    // stage K chunk (contiguous 32x896 bf16)
#if HAVE_TDM
    if (w == 0) tdm_load_2d_bf16(&Kg[baseK], &Ks[0], C_DIM, 32);
#else
    for (int idx = tid * 8; idx < 32 * C_DIM; idx += 256 * 8)
      async_copy16(&Kg[baseK + idx], &Ks[idx]);
#endif
    // stage V chunk transposed: Vst[d][key], two keys packed per b32 store
    for (int tsk = tid; tsk < 16 * 112; tsk += 256) {
      const int kp = (tsk / 112) << 1;   // even key
      const int d  = (tsk % 112) * 8;
      union { uint4 u; bf16 e[8]; } v0, v1;
      v0.u = *(const uint4*)&Vg[baseK + (size_t)kp * C_DIM + d];
      v1.u = *(const uint4*)&Vg[baseK + (size_t)(kp + 1) * C_DIM + d];
#pragma unroll
      for (int e = 0; e < 8; ++e) {
        union { unsigned u; bf16 b[2]; } p;
        p.b[0] = v0.e[e]; p.b[1] = v1.e[e];
        *(unsigned*)&Vst[(d + e) * 32 + kp] = p.u;
      }
    }
    for (int i = tid; i < 16 * 32; i += 256) Ssc[i] = 0.0f;
    async_wait();
    tensor_wait();
    __syncthreads();

    // S = Q * K^T : waves split the 28 d-steps
    v8f s0 = v8f_zero(), s1 = v8f_zero();
    for (int s = w; s < KSTEPS; s += 8) {
      const int kd = s * 32;
      const v16bf a  = load_a_frag(Qs + kd, C_DIM, lane);
      const v16bf b0 = load_b_frag(Ks + kd,              C_DIM, lane);
      const v16bf b1 = load_b_frag(Ks + 16 * C_DIM + kd, C_DIM, lane);
      s0 = wmma_bf16(a, b0, s0);
      s1 = wmma_bf16(a, b1, s1);
    }
    {
      const int rb = (lane >> 4) << 3, cl = lane & 15;
#pragma unroll
      for (int r = 0; r < 8; ++r) {
        atomicAdd(&Ssc[(rb + r) * 32 + cl],      s0[r]);
        atomicAdd(&Ssc[(rb + r) * 32 + cl + 16], s1[r]);
      }
    }
    __syncthreads();

    // online softmax (scale folded into Q)
    if (tid < 16) {
      const float m_old = mrow[tid];
      float mx = m_old;
#pragma unroll
      for (int c = 0; c < 32; ++c) mx = fmaxf(mx, Ssc[tid * 32 + c]);
      const float corr = __expf(m_old - mx);
      float sum = 0.0f;
#pragma unroll
      for (int c = 0; c < 32; ++c) {
        const float p = __expf(Ssc[tid * 32 + c] - mx);
        Ps[tid * 32 + c] = (bf16)p;
        sum += p;
      }
      mrow[tid] = mx;
      lrow[tid] = lrow[tid] * corr + sum;
      crow[tid] = corr;
    }
    __syncthreads();

    // rescale O, accumulate P*V on this wave's 112-wide slice
    {
      const int rb = (lane >> 4) << 3;
      float cr[8];
#pragma unroll
      for (int r = 0; r < 8; ++r) cr[r] = crow[rb + r];
      const v16bf a = load_a_frag(Ps, 32, lane);
#pragma unroll
      for (int t = 0; t < 7; ++t) {
#pragma unroll
        for (int r = 0; r < 8; ++r) oacc[t][r] *= cr[r];
        const v16bf bv = load_b_frag(Vst + (w * 112 + t * 16) * 32, 32, lane);
        oacc[t] = wmma_bf16(a, bv, oacc[t]);
      }
    }
  }

  __syncthreads();
  if (tid < 16) linv[tid] = 1.0f / lrow[tid];
  __syncthreads();
  {
    const int rb = (lane >> 4) << 3, cl = lane & 15;
#pragma unroll
    for (int t = 0; t < 7; ++t) {
      const int col = w * 112 + t * 16 + cl;
#pragma unroll
      for (int r = 0; r < 8; ++r) {
        const int row = q0 + rb + r;
        O[((size_t)b * N_TOK + row) * C_DIM + col] =
            (bf16)(oacc[t][r] * linv[rb + r]);
      }
    }
  }
}

// ---------------------------------------------------------------------------
// Kernel 3: out projection fused with mean pool.
// ---------------------------------------------------------------------------
__global__ __launch_bounds__(256) void proj_pool_kernel(
    const bf16* __restrict__ A, const float* __restrict__ Wp,
    float* __restrict__ pooled) {
  __shared__ bf16  As[128 * 40];
  __shared__ bf16  Bs[128 * 48];
  __shared__ float colsum[128];

  const int tid  = threadIdx.x;
  const int lane = tid & 31;
  const int w    = tid >> 5;
  const int wr   = w >> 1;
  const int wc   = w & 1;
  const int m0   = blockIdx.x * 128;
  const int n0   = blockIdx.y * 128;

  v8f acc[2][4];
#pragma unroll
  for (int i = 0; i < 2; ++i)
#pragma unroll
    for (int j = 0; j < 4; ++j) acc[i][j] = v8f_zero();

  const int arow  = tid >> 1;
  const int akoff = (tid & 1) * 16;

  for (int kt = 0; kt < KSTEPS; ++kt) {
    const int k0 = kt * 32;
    {
      const uint4* src =
          (const uint4*)(A + (size_t)(m0 + arow) * C_DIM + k0 + akoff);
      *(uint4*)&As[arow * 40 + akoff]     = src[0];
      *(uint4*)&As[arow * 40 + akoff + 8] = src[1];
    }
    stage_b_tile(Wp, Bs, k0, n0, tid);
    if (kt + 1 < KSTEPS)
      __builtin_prefetch(Wp + (size_t)(k0 + 32 + ((tid >> 4) << 1)) * C_DIM + n0 +
                             (tid & 15) * 8, 0, 1);
    __syncthreads();

    const v16bf a0 = load_a_frag(As + (wr * 32 + 0)  * 40, 40, lane);
    const v16bf a1 = load_a_frag(As + (wr * 32 + 16) * 40, 40, lane);
#pragma unroll
    for (int j = 0; j < 4; ++j) {
      const v16bf b = load_b_frag(Bs + (wc * 64 + j * 16) * 48, 48, lane);
      acc[0][j] = wmma_bf16(a0, b, acc[0][j]);
      acc[1][j] = wmma_bf16(a1, b, acc[1][j]);
    }
    __syncthreads();
  }

  if (tid < 128) colsum[tid] = 0.0f;
  __syncthreads();
  {
    const int cl = lane & 15;
#pragma unroll
    for (int i = 0; i < 2; ++i)
#pragma unroll
      for (int j = 0; j < 4; ++j) {
        const int c = wc * 64 + j * 16 + cl;
#pragma unroll
        for (int r = 0; r < 8; ++r) atomicAdd(&colsum[c], acc[i][j][r]);
      }
  }
  __syncthreads();
  if (tid < 128) {
    const int bb = m0 / N_TOK;  // 128-row tile never straddles a batch
    atomicAdd(&pooled[bb * C_DIM + n0 + tid], colsum[tid] * (1.0f / N_TOK));
  }
}

// ---------------------------------------------------------------------------
// Small kernels
// ---------------------------------------------------------------------------
__global__ void zero_pool_kernel(float* __restrict__ pooled) {
  const int gid = blockIdx.x * blockDim.x + threadIdx.x;
  if (gid < B_SZ * C_DIM) pooled[gid] = 0.0f;
}

__global__ void residual_mean_kernel(const float* __restrict__ x,
                                     const float* __restrict__ bp,
                                     float* __restrict__ pooled) {
  const int gid = blockIdx.x * blockDim.x + threadIdx.x;
  if (gid >= B_SZ * C_DIM) return;
  const int b = gid / C_DIM, c = gid % C_DIM;
  const float* p = x + (size_t)b * N_TOK * C_DIM + c;
  float s = 0.0f;
  for (int n = 0; n < N_TOK; ++n) s += p[(size_t)n * C_DIM];
  pooled[gid] += s * (1.0f / N_TOK) + bp[c];
}

__global__ void mlp1_kernel(const float* __restrict__ pooled,
                            const float* __restrict__ Wf1,
                            const float* __restrict__ bf1,
                            float* __restrict__ h) {
  const int gid = blockIdx.x * blockDim.x + threadIdx.x;
  if (gid >= B_SZ * C_DIM) return;
  const int b = gid / C_DIM, d = gid % C_DIM;
  float s = bf1[d];
  for (int c = 0; c < C_DIM; ++c)
    s += pooled[b * C_DIM + c] * Wf1[(size_t)c * C_DIM + d];
  h[gid] = fmaxf(s, 0.0f);
}

__global__ void mlp2_kernel(const float* __restrict__ h,
                            const float* __restrict__ Wf2,
                            const float* __restrict__ bf2,
                            float* __restrict__ out) {
  const int gid = blockIdx.x * blockDim.x + threadIdx.x;
  if (gid >= B_SZ * C_DIM) return;
  const int b = gid / C_DIM, e = gid % C_DIM;
  float s = bf2[e];
  for (int d = 0; d < C_DIM; ++d)
    s += h[b * C_DIM + d] * Wf2[(size_t)d * C_DIM + e];
  out[gid] = s;
}

// ---------------------------------------------------------------------------
extern "C" void kernel_launch(void* const* d_in, const int* in_sizes, int n_in,
                              void* d_out, int out_size, void* d_ws,
                              size_t ws_size, hipStream_t stream) {
  const float* x   = (const float*)d_in[0];
  const float* Wq  = (const float*)d_in[1];
  const float* Wk  = (const float*)d_in[2];
  const float* Wv  = (const float*)d_in[3];
  const float* Wp  = (const float*)d_in[4];
  const float* bp  = (const float*)d_in[5];
  const float* Wf1 = (const float*)d_in[6];
  const float* bf1 = (const float*)d_in[7];
  const float* Wf2 = (const float*)d_in[8];
  const float* bf2 = (const float*)d_in[9];

  const size_t SZ = (size_t)B_SZ * N_TOK * C_DIM;
  bf16* Qb = (bf16*)d_ws;
  bf16* Kb = Qb + SZ;
  bf16* Vb = Kb + SZ;
  bf16* Ob = Vb + SZ;
  float* pooled = (float*)(Ob + SZ);
  float* h      = pooled + B_SZ * C_DIM;

  qkv_gemm_kernel<<<dim3((B_SZ * N_TOK) / 128, C_DIM / 128, 3), 256, 0, stream>>>(
      x, Wq, Wk, Wv, Qb, Kb, Vb);
  flash_attn_kernel<<<dim3(N_TOK / 16, B_SZ), 256, 0, stream>>>(Qb, Kb, Vb, Ob);
  zero_pool_kernel<<<(B_SZ * C_DIM + 255) / 256, 256, 0, stream>>>(pooled);
  proj_pool_kernel<<<dim3((B_SZ * N_TOK) / 128, C_DIM / 128), 256, 0, stream>>>(
      Ob, Wp, pooled);
  residual_mean_kernel<<<(B_SZ * C_DIM + 255) / 256, 256, 0, stream>>>(x, bp, pooled);
  mlp1_kernel<<<(B_SZ * C_DIM + 255) / 256, 256, 0, stream>>>(pooled, Wf1, bf1, h);
  mlp2_kernel<<<(B_SZ * C_DIM + 255) / 256, 256, 0, stream>>>(h, Wf2, bf2,
                                                              (float*)d_out);
}
